// Mamba_12601434046490
// MI455X (gfx1250) — matrix operand order repaired and checked by reading, not compiled, not run on hardware
//
#include <hip/hip_runtime.h>
#include <hip/hip_bf16.h>
#include <stdint.h>

// ---- problem constants (from reference) ----
#define NLAY    4
#define DMODEL  512
#define DIN     768
#define DOUTC   40
#define KCONV   5
#define DINNER  1024
#define DSTATE  16
#define DCONV   4
#define DTRANK  32
#define BB      4
#define LLEN    1024
#define ROWS    (BB*LLEN)   // 4096
#define LPAD    (LLEN+4)    // 1028 (2 zero rows each side)

typedef __bf16 bf16_t;
typedef bf16_t v16bf __attribute__((ext_vector_type(16)));
typedef bf16_t v8bf  __attribute__((ext_vector_type(8)));
typedef float  v8f   __attribute__((ext_vector_type(8)));
typedef int    v4i_t __attribute__((ext_vector_type(4)));

// async global->LDS path (gfx1250): guarded so compile can't fail
#if defined(__has_builtin)
#  if __has_builtin(__builtin_amdgcn_global_load_async_to_lds_b128) && \
      __has_builtin(__builtin_amdgcn_s_wait_asynccnt)
#    define HAVE_ASYNC_LDS 1
#  endif
#endif
#ifndef HAVE_ASYNC_LDS
#  define HAVE_ASYNC_LDS 0
#endif

#define LAS __attribute__((address_space(3)))

__device__ __forceinline__ uint16_t f2bf(float f) {
    union { float f; uint32_t u; } v; v.f = f;
    uint32_t u = v.u;
    uint32_t r = (u + 0x7FFFu + ((u >> 16) & 1u)) >> 16;  // RNE
    return (uint16_t)r;
}

// =====================================================================
// WMMA GEMM:  C[M,N] (+)= A[M,K](bf16,lda) * W[N,K](bf16,ldw)^T
// Block: 256 thr / 8 waves. Wave tile 32x64 (2 A frags x 4 B frags,
// 8 wmma per K-step). Block tile 256x64.
// B tile (64n x 32k) staged in LDS, double buffered, loaded with
// async global->LDS (ASYNCcnt) when available; banks de-conflicted
// with a 16B row pad. A stays in registers from global (wave-private).
// grid.z batches A/C with element strides (padded-input conv).
// EPI: 0 = none, 1 = softplus(x + bias[n]) (dt_proj fusion).
// =====================================================================
#define BROW 80                        // LDS row stride bytes (64 data + 16 pad)
#define BTILE (64 * BROW)              // one B tile in LDS

template<bool ACC, int EPI>
__global__ __launch_bounds__(256) void gemm_bf16_wmma(
    const uint16_t* __restrict__ Abf, const uint16_t* __restrict__ Wbf,
    float* __restrict__ Cf, const float* __restrict__ bias,
    int M, int N, int K, int lda, int ldw, int ldc,
    long long strideA, long long strideC)
{
    __shared__ alignas(16) uint8_t sB[2 * BTILE];   // 10 KB, double buffered

    const int tid  = threadIdx.x;
    const int lane = tid & 31;
    const int wave = tid >> 5;
    const int m0 = blockIdx.y * 256 + wave * 32;
    const int n0 = blockIdx.x * 64;
    const uint16_t* A = Abf + (size_t)blockIdx.z * (size_t)strideA;
    float* C = Cf + (size_t)blockIdx.z * (size_t)strideC;

    const int lh   = lane >> 4;   // half select
    const int ll16 = lane & 15;

    // A frags: rows m0+ll16 and m0+16+ll16; elems0..15 <- K k+16*lh+0..15
    // (contiguous, matching B frag from LDS).
    const uint16_t* arow0 = A + (size_t)(m0 + ll16) * lda + lh * 16;
    const uint16_t* arow1 = arow0 + (size_t)16 * lda;

    // cooperative B-tile loader: thread -> (row, 16B chunk)
    const int bn = tid >> 2;                 // 0..63 (N row in tile)
    const int bc = tid & 3;                  // 16B chunk
    const uint16_t* wsrc = Wbf + (size_t)(n0 + bn) * ldw + bc * 8;
    uint8_t* sdst = sB + bn * BROW + bc * 16;

    // B frag source in LDS: column n0+t*16+ll16, 32B at K-half lh
    const uint8_t* sbrd = sB + ll16 * BROW + lh * 32;

    v8f acc[2][4] = {};
    const int KT = K / 32;

#if HAVE_ASYNC_LDS
    __builtin_amdgcn_global_load_async_to_lds_b128(
        (v4i_t*)wsrc, (LAS v4i_t*)sdst, 0, 0);
    for (int kt = 0; kt < KT; ++kt) {
        const int cur = kt & 1;
        const int k = kt * 32;
        __builtin_amdgcn_s_wait_asynccnt(0);   // own slice of buf[cur] done
        __syncthreads();                        // everyone's slice visible
        if (kt + 1 < KT)
            __builtin_amdgcn_global_load_async_to_lds_b128(
                (v4i_t*)(wsrc + (kt + 1) * 32),
                (LAS v4i_t*)(sdst + (1 - cur) * BTILE), 0, 0);
        v8bf alo0 = *(const v8bf*)(arow0 + k);
        v8bf ahi0 = *(const v8bf*)(arow0 + k + 8);
        v8bf alo1 = *(const v8bf*)(arow1 + k);
        v8bf ahi1 = *(const v8bf*)(arow1 + k + 8);
        v16bf a0 = __builtin_shufflevector(alo0, ahi0, 0,1,2,3,4,5,6,7,8,9,10,11,12,13,14,15);
        v16bf a1 = __builtin_shufflevector(alo1, ahi1, 0,1,2,3,4,5,6,7,8,9,10,11,12,13,14,15);
#pragma unroll
        for (int t = 0; t < 4; ++t) {
            v16bf b = *(const v16bf*)(sbrd + cur * BTILE + t * 16 * BROW);
            acc[0][t] = __builtin_amdgcn_wmma_f32_16x16x32_bf16(
                            false, a0, false, b, (short)0, acc[0][t], false, false);
            acc[1][t] = __builtin_amdgcn_wmma_f32_16x16x32_bf16(
                            false, a1, false, b, (short)0, acc[1][t], false, false);
        }
        __syncthreads();                        // done reading buf[cur]
    }
#else
    for (int kt = 0; kt < KT; ++kt) {
        const int k = kt * 32;
        uint4 wv = *(const uint4*)(wsrc + k);   // 16B of W
        __syncthreads();                        // prev tile fully read
        *(uint4*)sdst = wv;
        __syncthreads();
        v8bf alo0 = *(const v8bf*)(arow0 + k);
        v8bf ahi0 = *(const v8bf*)(arow0 + k + 8);
        v8bf alo1 = *(const v8bf*)(arow1 + k);
        v8bf ahi1 = *(const v8bf*)(arow1 + k + 8);
        v16bf a0 = __builtin_shufflevector(alo0, ahi0, 0,1,2,3,4,5,6,7,8,9,10,11,12,13,14,15);
        v16bf a1 = __builtin_shufflevector(alo1, ahi1, 0,1,2,3,4,5,6,7,8,9,10,11,12,13,14,15);
#pragma unroll
        for (int t = 0; t < 4; ++t) {
            v16bf b = *(const v16bf*)(sbrd + t * 16 * BROW);
            acc[0][t] = __builtin_amdgcn_wmma_f32_16x16x32_bf16(
                            false, a0, false, b, (short)0, acc[0][t], false, false);
            acc[1][t] = __builtin_amdgcn_wmma_f32_16x16x32_bf16(
                            false, a1, false, b, (short)0, acc[1][t], false, false);
        }
    }
#endif

    // D layout: VGPR v -> M = mbase + v ; N = n0 + t*16 + ll16
#pragma unroll
    for (int g = 0; g < 2; ++g) {
        const int mbase = m0 + g * 16 + lh * 8;
#pragma unroll
        for (int t = 0; t < 4; ++t) {
            const int n = n0 + t * 16 + ll16;
            float bv = (EPI == 1) ? bias[n] : 0.f;
#pragma unroll
            for (int v = 0; v < 8; ++v) {
                float* cp = C + (size_t)(mbase + v) * ldc + n;
                float val = acc[g][t][v];
                if (ACC) val += *cp;
                if (EPI == 1) {
                    val += bv;
                    val = (val > 20.f) ? val : log1pf(__expf(val));  // softplus
                }
                *cp = val;
            }
        }
    }
}

// ---- prep kernels -------------------------------------------------
__global__ void f32_to_bf16_kernel(const float* __restrict__ s,
                                   uint16_t* __restrict__ d, int n) {
    int i = blockIdx.x * blockDim.x + threadIdx.x;
    if (i < n) d[i] = f2bf(s[i]);
}

__global__ void negexp_kernel(const float* __restrict__ s,
                              float* __restrict__ d, int n) {
    int i = blockIdx.x * blockDim.x + threadIdx.x;
    if (i < n) d[i] = -__expf(s[i]);
}

// x[B,DIN,L] -> xT_bf16[B, LPAD, DIN] with 2 zero rows of padding each side
__global__ void build_xT_kernel(const float* __restrict__ x,
                                uint16_t* __restrict__ xT) {
    int idx = blockIdx.x * blockDim.x + threadIdx.x;
    if (idx >= BB * LPAD * DIN) return;
    int d = idx % DIN;
    int r = (idx / DIN) % LPAD;
    int b = idx / (DIN * LPAD);
    float v = 0.f;
    int l = r - 2;
    if (l >= 0 && l < LLEN) v = x[((size_t)b * DIN + d) * LLEN + l];
    xT[idx] = f2bf(v);
}

// in_conv_w[M,DIN,5] -> wk[k][M,DIN] bf16 (per-tap weight matrices)
__global__ void prep_wconv_kernel(const float* __restrict__ w,
                                  uint16_t* __restrict__ wk) {
    int idx = blockIdx.x * blockDim.x + threadIdx.x;
    if (idx >= KCONV * DMODEL * DIN) return;
    int d = idx % DIN;
    int m = (idx / DIN) % DMODEL;
    int k = idx / (DIN * DMODEL);
    wk[idx] = f2bf(w[((size_t)m * DIN + d) * KCONV + k]);
}

// h = (h + bias[m]) * (l < len[b])
__global__ void bias_mask_kernel(float* __restrict__ h,
                                 const float* __restrict__ bias,
                                 const int* __restrict__ lengths) {
    int idx = blockIdx.x * blockDim.x + threadIdx.x;
    if (idx >= ROWS * DMODEL) return;
    int m = idx & (DMODEL - 1);
    int row = idx >> 9;
    int b = row >> 10, l = row & (LLEN - 1);
    float v = h[idx] + bias[m];
    h[idx] = (l < lengths[b]) ? v : 0.f;
}

// residual (+)= h ; hn = LN(residual)*w + b  -> bf16
__global__ __launch_bounds__(256) void ln_kernel(
    const float* __restrict__ h, float* __restrict__ res,
    const float* __restrict__ w, const float* __restrict__ bc,
    uint16_t* __restrict__ hnbf, int add)
{
    __shared__ float s1[256], s2[256];
    int row = blockIdx.x, c = threadIdx.x;
    size_t base = (size_t)row * DMODEL;
    float a0 = h[base + c];
    float a1 = h[base + c + 256];
    if (add) { a0 += res[base + c]; a1 += res[base + c + 256]; }
    res[base + c] = a0; res[base + c + 256] = a1;
    s1[c] = a0 + a1; s2[c] = a0 * a0 + a1 * a1;
    __syncthreads();
    for (int off = 128; off > 0; off >>= 1) {
        if (c < off) { s1[c] += s1[c + off]; s2[c] += s2[c + off]; }
        __syncthreads();
    }
    float mean = s1[0] * (1.f / DMODEL);
    float var  = s2[0] * (1.f / DMODEL) - mean * mean;
    float rs = rsqrtf(var + 1e-5f);
    hnbf[base + c]       = f2bf((a0 - mean) * rs * w[c] + bc[c]);
    hnbf[base + c + 256] = f2bf((a1 - mean) * rs * w[c + 256] + bc[c + 256]);
}

// depthwise causal conv (pad 3,0) + bias + SiLU on xi = xz[:, :DINNER]
__global__ void dwconv_silu_kernel(const float* __restrict__ xz,
    const float* __restrict__ cw, const float* __restrict__ cb,
    float* __restrict__ xc, uint16_t* __restrict__ xcbf)
{
    int idx = blockIdx.x * blockDim.x + threadIdx.x;
    if (idx >= ROWS * DINNER) return;
    int d = idx & (DINNER - 1);
    int row = idx >> 10;
    int b = row >> 10, l = row & (LLEN - 1);
    float acc = cb[d];
#pragma unroll
    for (int k = 0; k < DCONV; ++k) {
        int ls = l - (DCONV - 1) + k;
        if (ls >= 0)
            acc += xz[((size_t)(b * LLEN + ls)) * 2 * DINNER + d] * cw[d * DCONV + k];
    }
    float s = acc / (1.f + __expf(-acc));
    xc[idx] = s;
    xcbf[idx] = f2bf(s);
}

// sequential selective scan: one thread per (b, d) channel, 16 states in regs
__global__ __launch_bounds__(256) void selective_scan_kernel(
    const float* __restrict__ xc, const float* __restrict__ dt,
    const float* __restrict__ xdbl, const float* __restrict__ xz,
    const float* __restrict__ Aneg, const float* __restrict__ Dv,
    uint16_t* __restrict__ ybf)
{
    int t = blockIdx.x * blockDim.x + threadIdx.x;
    if (t >= BB * DINNER) return;
    int b = t >> 10;
    int d = t & (DINNER - 1);
    float Ar[DSTATE], hs[DSTATE];
#pragma unroll
    for (int n = 0; n < DSTATE; ++n) { Ar[n] = Aneg[d * DSTATE + n]; hs[n] = 0.f; }
    float Dd = Dv[d];
    for (int l = 0; l < LLEN; ++l) {
        size_t row = (size_t)b * LLEN + l;
        float dtv = dt[row * DINNER + d];
        float u   = xc[row * DINNER + d];
        const float* bd = xdbl + row * (DTRANK + 2 * DSTATE);
        float y = 0.f;
#pragma unroll
        for (int n = 0; n < DSTATE; ++n) {
            float dA = __expf(dtv * Ar[n]);
            hs[n] = dA * hs[n] + dtv * bd[DTRANK + n] * u;
            y += hs[n] * bd[DTRANK + DSTATE + n];
        }
        y += u * Dd;
        float z = xz[row * 2 * DINNER + DINNER + d];
        float sil = z / (1.f + __expf(-z));
        ybf[row * DINNER + d] = f2bf(y * sil);
    }
}

// out conv (512 -> 40, K=5, SAME) + bias, masked
__global__ void out_conv_kernel(const float* __restrict__ h,
    const float* __restrict__ w, const float* __restrict__ bias,
    const int* __restrict__ lengths, float* __restrict__ out)
{
    int idx = blockIdx.x * blockDim.x + threadIdx.x;
    if (idx >= BB * DOUTC * LLEN) return;
    int l = idx % LLEN;
    int o = (idx / LLEN) % DOUTC;
    int b = idx / (LLEN * DOUTC);
    if (l >= lengths[b]) { out[idx] = 0.f; return; }
    float acc = bias[o];
    const float* wo = w + (size_t)o * DMODEL * KCONV;
    for (int m = 0; m < DMODEL; ++m) {
#pragma unroll
        for (int k = 0; k < KCONV; ++k) {
            int ls = l + k - 2;
            if (ls >= 0 && ls < LLEN)
                acc += h[((size_t)b * LLEN + ls) * DMODEL + m] * wo[m * KCONV + k];
        }
    }
    out[idx] = acc;
}

// =====================================================================
extern "C" void kernel_launch(void* const* d_in, const int* in_sizes, int n_in,
                              void* d_out, int out_size, void* d_ws, size_t ws_size,
                              hipStream_t stream) {
    const float* x          = (const float*)d_in[0];
    const int*   lengths    = (const int*)d_in[1];
    const float* in_conv_w  = (const float*)d_in[2];
    const float* in_conv_b  = (const float*)d_in[3];
    const float* out_conv_w = (const float*)d_in[4];
    const float* out_conv_b = (const float*)d_in[5];
    const float* in_proj_w  = (const float*)d_in[6];
    const float* conv_w     = (const float*)d_in[7];
    const float* conv_b     = (const float*)d_in[8];
    const float* x_proj_w   = (const float*)d_in[9];
    const float* dt_proj_w  = (const float*)d_in[10];
    const float* dt_proj_b  = (const float*)d_in[11];
    const float* A_log      = (const float*)d_in[12];
    const float* Dp         = (const float*)d_in[13];
    const float* out_proj_w = (const float*)d_in[14];
    const float* norm_w     = (const float*)d_in[15];
    const float* norm_b     = (const float*)d_in[16];
    float* out = (float*)d_out;

    // ---- workspace layout (256B aligned) ----
    char* base = (char*)d_ws;
    size_t off = 0;
    auto alloc = [&](size_t bytes) { size_t o = off; off += (bytes + 255) & ~(size_t)255; return o; };
    const int DXP = DTRANK + 2 * DSTATE;  // 64

    uint16_t* xT    = (uint16_t*)(base + alloc((size_t)BB * LPAD * DIN * 2));
    uint16_t* wck   = (uint16_t*)(base + alloc((size_t)KCONV * DMODEL * DIN * 2));
    uint16_t* w_in  = (uint16_t*)(base + alloc((size_t)NLAY * 2 * DINNER * DMODEL * 2));
    uint16_t* w_xp  = (uint16_t*)(base + alloc((size_t)NLAY * DXP * DINNER * 2));
    uint16_t* w_dt  = (uint16_t*)(base + alloc((size_t)NLAY * DINNER * DTRANK * 2));
    uint16_t* w_out = (uint16_t*)(base + alloc((size_t)NLAY * DMODEL * DINNER * 2));
    float*    Aneg  = (float*)   (base + alloc((size_t)NLAY * DINNER * DSTATE * 4));
    float*    hbuf  = (float*)   (base + alloc((size_t)ROWS * DMODEL * 4));
    float*    res   = (float*)   (base + alloc((size_t)ROWS * DMODEL * 4));
    uint16_t* hnbf  = (uint16_t*)(base + alloc((size_t)ROWS * DMODEL * 2));
    float*    xz    = (float*)   (base + alloc((size_t)ROWS * 2 * DINNER * 4));
    float*    xc    = (float*)   (base + alloc((size_t)ROWS * DINNER * 4));
    uint16_t* xcbf  = (uint16_t*)(base + alloc((size_t)ROWS * DINNER * 2));
    float*    xdbl  = (float*)   (base + alloc((size_t)ROWS * DXP * 4));
    uint16_t* xdblbf= (uint16_t*)(base + alloc((size_t)ROWS * DXP * 2));
    float*    dtbuf = (float*)   (base + alloc((size_t)ROWS * DINNER * 4));
    uint16_t* ybf   = (uint16_t*)(base + alloc((size_t)ROWS * DINNER * 2));
    (void)ws_size; (void)n_in; (void)in_sizes; (void)out_size;

    auto blk = [](int n) { return (n + 255) / 256; };

    // ---- weight prep ----
    f32_to_bf16_kernel<<<blk(NLAY*2*DINNER*DMODEL), 256, 0, stream>>>(in_proj_w,  w_in,  NLAY*2*DINNER*DMODEL);
    f32_to_bf16_kernel<<<blk(NLAY*DXP*DINNER),      256, 0, stream>>>(x_proj_w,   w_xp,  NLAY*DXP*DINNER);
    f32_to_bf16_kernel<<<blk(NLAY*DINNER*DTRANK),   256, 0, stream>>>(dt_proj_w,  w_dt,  NLAY*DINNER*DTRANK);
    f32_to_bf16_kernel<<<blk(NLAY*DMODEL*DINNER),   256, 0, stream>>>(out_proj_w, w_out, NLAY*DMODEL*DINNER);
    negexp_kernel<<<blk(NLAY*DINNER*DSTATE), 256, 0, stream>>>(A_log, Aneg, NLAY*DINNER*DSTATE);
    build_xT_kernel<<<blk(BB*LPAD*DIN), 256, 0, stream>>>(x, xT);
    prep_wconv_kernel<<<blk(KCONV*DMODEL*DIN), 256, 0, stream>>>(in_conv_w, wck);

    // ---- in_conv as 5 shifted batched GEMMs: h[b,l,m] = sum_k xT[b,l+k,:]*wk[k][m,:]
    {
        dim3 g(DMODEL / 64, LLEN / 256, BB);
        long long sA = (long long)LPAD * DIN, sC = (long long)LLEN * DMODEL;
        gemm_bf16_wmma<false, 0><<<g, 256, 0, stream>>>(
            xT + 0 * DIN, wck + 0 * (size_t)DMODEL * DIN, hbuf, nullptr,
            LLEN, DMODEL, DIN, DIN, DIN, DMODEL, sA, sC);
        for (int k = 1; k < KCONV; ++k)
            gemm_bf16_wmma<true, 0><<<g, 256, 0, stream>>>(
                xT + k * DIN, wck + (size_t)k * DMODEL * DIN, hbuf, nullptr,
                LLEN, DMODEL, DIN, DIN, DIN, DMODEL, sA, sC);
    }
    bias_mask_kernel<<<blk(ROWS * DMODEL), 256, 0, stream>>>(hbuf, in_conv_b, lengths);

    // ---- layers ----
    for (int layer = 0; layer < NLAY; ++layer) {
        ln_kernel<<<ROWS, 256, 0, stream>>>(hbuf, res,
            norm_w + layer * DMODEL, norm_b + layer * DMODEL, hnbf, layer > 0);

        // xz = hn @ in_proj_w^T   [4096,512]x[2048,512]^T
        gemm_bf16_wmma<false, 0><<<dim3(2 * DINNER / 64, ROWS / 256, 1), 256, 0, stream>>>(
            hnbf, w_in + (size_t)layer * 2 * DINNER * DMODEL, xz, nullptr,
            ROWS, 2 * DINNER, DMODEL, DMODEL, DMODEL, 2 * DINNER, 0, 0);

        dwconv_silu_kernel<<<blk(ROWS * DINNER), 256, 0, stream>>>(
            xz, conv_w + (size_t)layer * DINNER * DCONV, conv_b + layer * DINNER, xc, xcbf);

        // xdbl = xc @ x_proj_w^T  [4096,1024]x[64,1024]^T
        gemm_bf16_wmma<false, 0><<<dim3(DXP / 64, ROWS / 256, 1), 256, 0, stream>>>(
            xcbf, w_xp + (size_t)layer * DXP * DINNER, xdbl, nullptr,
            ROWS, DXP, DINNER, DINNER, DINNER, DXP, 0, 0);

        f32_to_bf16_kernel<<<blk(ROWS * DXP), 256, 0, stream>>>(xdbl, xdblbf, ROWS * DXP);

        // dt = softplus(xdbl[:, :32] @ dt_proj_w^T + dt_b)  [4096,32]x[1024,32]^T
        gemm_bf16_wmma<false, 1><<<dim3(DINNER / 64, ROWS / 256, 1), 256, 0, stream>>>(
            xdblbf, w_dt + (size_t)layer * DINNER * DTRANK, dtbuf,
            dt_proj_b + layer * DINNER,
            ROWS, DINNER, DTRANK, DXP, DTRANK, DINNER, 0, 0);

        selective_scan_kernel<<<blk(BB * DINNER), 256, 0, stream>>>(
            xc, dtbuf, xdbl, xz, Aneg + (size_t)layer * DINNER * DSTATE,
            Dp + layer * DINNER, ybf);

        // h = y @ out_proj_w^T  [4096,1024]x[512,1024]^T
        gemm_bf16_wmma<false, 0><<<dim3(DMODEL / 64, ROWS / 256, 1), 256, 0, stream>>>(
            ybf, w_out + (size_t)layer * DMODEL * DINNER, hbuf, nullptr,
            ROWS, DMODEL, DINNER, DINNER, DINNER, DMODEL, 0, 0);
    }

    out_conv_kernel<<<blk(BB * DOUTC * LLEN), 256, 0, stream>>>(
        hbuf, out_conv_w, out_conv_b, lengths, out);
}
